// SpatialSelfAttention_7361573945891
// MI455X (gfx1250) — matrix operand order repaired and checked by reading, not compiled, not run on hardware
//
#include <hip/hip_runtime.h>

typedef __attribute__((ext_vector_type(16))) __bf16 v16bf;
typedef __attribute__((ext_vector_type(8)))  __bf16 v8bf;
typedef __attribute__((ext_vector_type(8)))  float  v8f;

union ABf { v16bf v; v8bf h[2]; };

#define WMMA_BF16(A, B, Cacc) \
  __builtin_amdgcn_wmma_f32_16x16x32_bf16(false, (A), false, (B), (short)0, (Cacc), false, false)

// Problem constants
#define BATCH 4
#define CH    256      // channels
#define NS    4096     // spatial (64*64)
#define NGRP  16
#define CPG   16       // channels per group

// ---------------------------------------------------------------------------
// Kernel 1: GroupNorm statistics. One block per (b, group).
// Channels of a group are contiguous in memory: x[b][g*16 .. g*16+15][*].
// ---------------------------------------------------------------------------
__global__ void gn_stats_kernel(const float* __restrict__ x, float* __restrict__ stats) {
  __shared__ float sh_s[256];
  __shared__ float sh_q[256];
  int bg = blockIdx.x;
  const float* base = x + (size_t)bg * (CPG * NS);
  float s = 0.f, q = 0.f;
  for (int i = threadIdx.x; i < CPG * NS; i += 256) {
    float v = base[i];
    s += v;
    q += v * v;
  }
  sh_s[threadIdx.x] = s;
  sh_q[threadIdx.x] = q;
  __syncthreads();
  for (int off = 128; off > 0; off >>= 1) {
    if ((int)threadIdx.x < off) {
      sh_s[threadIdx.x] += sh_s[threadIdx.x + off];
      sh_q[threadIdx.x] += sh_q[threadIdx.x + off];
    }
    __syncthreads();
  }
  if (threadIdx.x == 0) {
    float mean = sh_s[0] * (1.f / 65536.f);
    float var  = sh_q[0] * (1.f / 65536.f) - mean * mean;
    stats[bg * 2 + 0] = mean;
    stats[bg * 2 + 1] = rsqrtf(var + 1e-6f);
  }
}

// ---------------------------------------------------------------------------
// Kernel 2: normalize + transpose to hnT[b][n][c] (bf16). 64x64 LDS tiles so
// both global read (along n) and global write (along c) are coalesced.
// Grid: (NS/64, CH/64, BATCH), block 256.
// ---------------------------------------------------------------------------
__global__ void gn_norm_t_kernel(const float* __restrict__ x,
                                 const float* __restrict__ gamma,
                                 const float* __restrict__ beta,
                                 const float* __restrict__ stats,
                                 __bf16* __restrict__ hnT) {
  __shared__ float tile[64][65];
  int n0 = blockIdx.x * 64;
  int c0 = blockIdx.y * 64;
  int b  = blockIdx.z;
  int nl  = threadIdx.x & 63;
  int cl0 = threadIdx.x >> 6;   // 0..3
#pragma unroll
  for (int i = 0; i < 16; i++) {
    int cl = cl0 * 16 + i;
    int c  = c0 + cl;
    float mean = stats[(b * NGRP + (c >> 4)) * 2 + 0];
    float rstd = stats[(b * NGRP + (c >> 4)) * 2 + 1];
    float v = x[((size_t)(b * CH + c)) * NS + n0 + nl];
    tile[cl][nl] = (v - mean) * rstd * gamma[c] + beta[c];
  }
  __syncthreads();
  int cl  = threadIdx.x & 63;
  int nl0 = threadIdx.x >> 6;
#pragma unroll
  for (int i = 0; i < 16; i++) {
    int nn = nl0 * 16 + i;
    hnT[((size_t)(b * NS + n0 + nn)) * CH + c0 + cl] = (__bf16)tile[cl][nn];
  }
}

// ---------------------------------------------------------------------------
// Kernel 3: convert the four 256x256 weight matrices to bf16 (row-major [O,C]).
// Grid: (256, 4), block 256.
// ---------------------------------------------------------------------------
__global__ void cvt_w_kernel(const float* __restrict__ wq, const float* __restrict__ wk,
                             const float* __restrict__ wv, const float* __restrict__ wp,
                             __bf16* __restrict__ wb) {
  int m = blockIdx.y;
  int i = blockIdx.x * 256 + threadIdx.x;
  const float* src = (m == 0) ? wq : (m == 1) ? wk : (m == 2) ? wv : wp;
  wb[(size_t)m * (CH * CH) + i] = (__bf16)src[i];
}

// ---------------------------------------------------------------------------
// Kernel 4: Q/K/V 1x1 conv GEMM. One wave per (b, 16-row n tile, 16-col o
// tile); A fragments (hnT rows) shared across the three outputs.
//   q,k -> [B,N,C] bf16 ; v -> vT [B,C,N] bf16 (contiguous per-lane stores).
// ---------------------------------------------------------------------------
__global__ __launch_bounds__(128) void qkv_kernel(const __bf16* __restrict__ hnT,
                                                  const __bf16* __restrict__ wb,
                                                  const float* __restrict__ bq,
                                                  const float* __restrict__ bk,
                                                  const float* __restrict__ bv,
                                                  __bf16* __restrict__ qm,
                                                  __bf16* __restrict__ km,
                                                  __bf16* __restrict__ vT) {
  int wave = blockIdx.x * 4 + (threadIdx.x >> 5);
  int b  = wave >> 12;           // 4096 waves per batch
  int nt = (wave >> 4) & 255;
  int ot = wave & 15;
  int n0 = nt * 16, o0 = ot * 16;
  int L  = threadIdx.x & 31;
  int m_ = L & 15;
  int h  = L >> 4;

  const __bf16* arow = hnT + ((size_t)(b * NS) + n0 + m_) * CH;
  const __bf16* wqb  = wb;
  const __bf16* wkb  = wb + CH * CH;
  const __bf16* wvb  = wb + 2 * CH * CH;

  v8f aq = {}, ak = {}, av = {};
#pragma unroll
  for (int kk = 0; kk < 8; kk++) {
    int k0 = kk * 32;
    ABf a;
    a.h[0] = *(const v8bf*)(arow + k0 + 16 * h);
    a.h[1] = *(const v8bf*)(arow + k0 + 16 * h + 8);
    v16bf bqf = *(const v16bf*)(wqb + (size_t)(o0 + m_) * CH + k0 + 16 * h);
    v16bf bkf = *(const v16bf*)(wkb + (size_t)(o0 + m_) * CH + k0 + 16 * h);
    v16bf bvf = *(const v16bf*)(wvb + (size_t)(o0 + m_) * CH + k0 + 16 * h);
    aq = WMMA_BF16(a.v, bqf, aq);
    ak = WMMA_BF16(a.v, bkf, ak);
    av = WMMA_BF16(a.v, bvf, av);
  }

  int o = o0 + m_;
  float bqs = bq[o], bks = bk[o], bvs = bv[o];

#pragma unroll
  for (int r = 0; r < 8; r++) {
    int n = n0 + r + 8 * h;
    qm[((size_t)(b * NS) + n) * CH + o] = (__bf16)(aq[r] + bqs);
    km[((size_t)(b * NS) + n) * CH + o] = (__bf16)(ak[r] + bks);
  }
  // v transposed: per-lane contiguous 8 bf16 along n
  v8bf pv;
#pragma unroll
  for (int r = 0; r < 8; r++) pv[r] = (__bf16)(av[r] + bvs);
  *(v8bf*)(vT + ((size_t)(b * CH) + o) * NS + n0 + 8 * h) = pv;
}

// ---------------------------------------------------------------------------
// Kernel 5: fused flash attention + projection + residual.
// Grid: (NS/64, BATCH), block 128 (4 waves). Wave w owns query rows
// i0 = blk*64 + w*16 and a 16x256 f32 accumulator in VGPRs.
// ---------------------------------------------------------------------------
__global__ __launch_bounds__(128) void attn_kernel(const __bf16* __restrict__ qm,
                                                   const __bf16* __restrict__ km,
                                                   const __bf16* __restrict__ vT,
                                                   const __bf16* __restrict__ wpb,
                                                   const float* __restrict__ bp,
                                                   const float* __restrict__ x,
                                                   float* __restrict__ out) {
  // per-wave private LDS scratch (same-wave LDS ops are in-order; no barriers)
  __shared__ __bf16 ldsP[4][16 * 40];    // P tile 16x32, row stride 40 (16B aligned)
  __shared__ __bf16 ldsH[4][16 * 272];   // H tile 16x256, row stride 272

  int w  = threadIdx.x >> 5;
  int L  = threadIdx.x & 31;
  int m_ = L & 15;
  int h  = L >> 4;
  int b  = blockIdx.y;
  int i0 = blockIdx.x * 64 + w * 16;

  __bf16* myP = ldsP[w];
  __bf16* myH = ldsH[w];

  v8f oa[16];
#pragma unroll
  for (int f = 0; f < 16; f++) oa[f] = (v8f){};
  float rmax[8], rsum[8];
#pragma unroll
  for (int r = 0; r < 8; r++) { rmax[r] = -__builtin_inff(); rsum[r] = 0.f; }

  const __bf16* qrow = qm + ((size_t)(b * NS) + i0 + m_) * CH;
  const float SCALE = 0.0625f;          // 256^-0.5
  const float L2E   = 1.44269504f;

#pragma unroll 1
  for (int jt = 0; jt < NS / 32; jt++) {
    int j0 = jt * 32;
    // ---- S = Q * K^T for a 16x32 j-slab (two 16x16 subtiles) ----
    v8f s0 = {}, s1 = {};
    const __bf16* k0p = km + ((size_t)(b * NS) + j0 + m_) * CH;
    const __bf16* k1p = k0p + 16 * CH;
#pragma unroll
    for (int kk = 0; kk < 8; kk++) {
      int kc = kk * 32;
      ABf a;
      a.h[0] = *(const v8bf*)(qrow + kc + 16 * h);
      a.h[1] = *(const v8bf*)(qrow + kc + 16 * h + 8);
      v16bf bk0 = *(const v16bf*)(k0p + kc + 16 * h);
      v16bf bk1 = *(const v16bf*)(k1p + kc + 16 * h);
      s0 = WMMA_BF16(a.v, bk0, s0);
      s1 = WMMA_BF16(a.v, bk1, s1);
    }
    // ---- online softmax (rows live in 16-lane halves) ----
    float corr[8];
#pragma unroll
    for (int r = 0; r < 8; r++) {
      float a0 = s0[r] * SCALE;
      float a1 = s1[r] * SCALE;
      float t = fmaxf(a0, a1);
      t = fmaxf(t, __shfl_xor(t, 1, 32));
      t = fmaxf(t, __shfl_xor(t, 2, 32));
      t = fmaxf(t, __shfl_xor(t, 4, 32));
      t = fmaxf(t, __shfl_xor(t, 8, 32));
      float mn = fmaxf(rmax[r], t);
      corr[r] = __builtin_amdgcn_exp2f((rmax[r] - mn) * L2E);
      float p0 = __builtin_amdgcn_exp2f((a0 - mn) * L2E);
      float p1 = __builtin_amdgcn_exp2f((a1 - mn) * L2E);
      float ps = p0 + p1;
      ps += __shfl_xor(ps, 1, 32);
      ps += __shfl_xor(ps, 2, 32);
      ps += __shfl_xor(ps, 4, 32);
      ps += __shfl_xor(ps, 8, 32);
      rsum[r] = rsum[r] * corr[r] + ps;
      rmax[r] = mn;
      // write P (D layout) into LDS
      myP[(r + 8 * h) * 40 + m_]      = (__bf16)p0;
      myP[(r + 8 * h) * 40 + 16 + m_] = (__bf16)p1;
    }
    // rescale accumulator
#pragma unroll
    for (int f = 0; f < 16; f++)
#pragma unroll
      for (int r = 0; r < 8; r++) oa[f][r] *= corr[r];
    // ---- O += P * V  (P re-read in A layout from LDS) ----
    ABf pa;
    pa.h[0] = *(const v8bf*)(myP + m_ * 40 + 16 * h);
    pa.h[1] = *(const v8bf*)(myP + m_ * 40 + 16 * h + 8);
#pragma unroll
    for (int f = 0; f < 16; f++) {
      v16bf vb = *(const v16bf*)(vT + ((size_t)(b * CH) + f * 16 + m_) * NS + j0 + 16 * h);
      oa[f] = WMMA_BF16(pa.v, vb, oa[f]);
    }
  }

  // ---- epilogue: normalize, bounce H through LDS, fused projection ----
  float inv[8];
#pragma unroll
  for (int r = 0; r < 8; r++) inv[r] = 1.0f / rsum[r];
#pragma unroll
  for (int f = 0; f < 16; f++)
#pragma unroll
    for (int r = 0; r < 8; r++)
      myH[(r + 8 * h) * 272 + f * 16 + m_] = (__bf16)(oa[f][r] * inv[r]);

  ABf ha[8];
#pragma unroll
  for (int kk = 0; kk < 8; kk++) {
    ha[kk].h[0] = *(const v8bf*)(myH + m_ * 272 + kk * 32 + 16 * h);
    ha[kk].h[1] = *(const v8bf*)(myH + m_ * 272 + kk * 32 + 16 * h + 8);
  }

#pragma unroll 1
  for (int f = 0; f < 16; f++) {
    v8f y = {};
#pragma unroll
    for (int kk = 0; kk < 8; kk++) {
      v16bf bw = *(const v16bf*)(wpb + (size_t)(f * 16 + m_) * CH + kk * 32 + 16 * h);
      y = WMMA_BF16(ha[kk].v, bw, y);
    }
    int o = f * 16 + m_;
    float bb = bp[o];
    size_t base = ((size_t)(b * CH) + o) * NS + i0 + 8 * h;
#pragma unroll
    for (int r = 0; r < 8; r++) out[base + r] = x[base + r] + y[r] + bb;
  }
}

// ---------------------------------------------------------------------------
// Host launcher
// ---------------------------------------------------------------------------
extern "C" void kernel_launch(void* const* d_in, const int* in_sizes, int n_in,
                              void* d_out, int out_size, void* d_ws, size_t ws_size,
                              hipStream_t stream) {
  const float* x     = (const float*)d_in[0];
  const float* gamma = (const float*)d_in[1];
  const float* beta  = (const float*)d_in[2];
  const float* wq    = (const float*)d_in[3];
  const float* bq    = (const float*)d_in[4];
  const float* wk    = (const float*)d_in[5];
  const float* bk    = (const float*)d_in[6];
  const float* wv    = (const float*)d_in[7];
  const float* bv    = (const float*)d_in[8];
  const float* wp    = (const float*)d_in[9];
  const float* bp    = (const float*)d_in[10];
  float* out = (float*)d_out;

  char* ws = (char*)d_ws;
  const size_t MAT = (size_t)BATCH * NS * CH;       // elements per bf16 matrix
  __bf16* hnT = (__bf16*)ws;                         // [B,N,C]
  __bf16* wb  = hnT + MAT;                           // 4x 256x256 weights
  __bf16* qm  = wb + (size_t)4 * CH * CH;            // [B,N,C]
  __bf16* km  = qm + MAT;                            // [B,N,C]
  __bf16* vT  = km + MAT;                            // [B,C,N]
  float* stats = (float*)(vT + MAT);                 // 64 x (mean, rstd)

  gn_stats_kernel<<<BATCH * NGRP, 256, 0, stream>>>(x, stats);
  gn_norm_t_kernel<<<dim3(NS / 64, CH / 64, BATCH), 256, 0, stream>>>(x, gamma, beta, stats, hnT);
  cvt_w_kernel<<<dim3(CH * CH / 256, 4), 256, 0, stream>>>(wq, wk, wv, wp, wb);
  qkv_kernel<<<(BATCH * (NS / 16) * (CH / 16)) / 4, 128, 0, stream>>>(hnT, wb, bq, bk, bv, qm, km, vT);
  attn_kernel<<<dim3(NS / 64, BATCH), 128, 0, stream>>>(qm, km, vT, wb + (size_t)3 * CH * CH, bp, x, out);
}
